// HSCATB_76716705841257
// MI455X (gfx1250) — matrix operand order repaired and checked by reading, not compiled, not run on hardware
//
#include <hip/hip_runtime.h>
#include <cstdint>
#include <cstddef>

// ---------------------------------------------------------------------------
// HSCATB fused block for MI455X (gfx1250, wave32, WMMA bf16 16x16x32).
//
// GEMM-heavy ops (qkv, proj, fc1, fc2, global attention) use
// v_wmma_f32_16x16x32_bf16 with a contiguous k = 16*half + s order in both A and B fragments.
// All GEMM activations are staged as bf16 in [n][channel] (K-contiguous)
// layout so A/B fragment loads are contiguous 32-bit bf16 pairs.
// GEMMs are register-blocked 32x64 per wave (8 WMMAs / k-step) for ~21 FLOP/B
// from L2 instead of ~8 with a single 16x16 tile.
// ---------------------------------------------------------------------------

typedef __bf16 bf16_t;
typedef __attribute__((ext_vector_type(16))) __bf16 v16bf;
typedef __attribute__((ext_vector_type(8)))  float  v8f;

#define BATCH   8
#define CHANS   256
#define SIDE    32
#define NPIX    1024      // SIDE*SIDE
#define NHEADS  8
#define HDIM    32        // CHANS / NHEADS
#define HIDDIM  1024      // 4*CHANS
#define K2DIM   2048      // 2*HIDDIM
#define QKV_LD  768       // 3*CHANS

__device__ __forceinline__ v8f wmma_bf16(v16bf a, v16bf b, v8f c) {
  return __builtin_amdgcn_wmma_f32_16x16x32_bf16(false, a, false, b, (short)0, c,
                                                 false, false);
}

// A fragment (16x32 bf16). lane: row = lane&15 for both halves.
// half 0: VGPR0..7 = K 0..15 pairs
// half 1: VGPR0..7 = K 16..31 pairs
__device__ __forceinline__ v16bf load_a_frag(const bf16_t* row, int hlf) {
  v16bf a;
#pragma unroll
  for (int j = 0; j < 4; ++j) {
    int k = hlf * 16 + 2 * j;
    a[2 * j]     = row[k];
    a[2 * j + 1] = row[k + 1];
  }
#pragma unroll
  for (int j = 0; j < 4; ++j) {
    int k = hlf * 16 + 8 + 2 * j;
    a[8 + 2 * j]     = row[k];
    a[8 + 2 * j + 1] = row[k + 1];
  }
  return a;
}

// B fragment (32x16 bf16). lane: col = lane&15; lanes 0-15 hold K 0..15,
// lanes 16-31 hold K 16..31. row pointer must be pre-offset by half*16.
__device__ __forceinline__ v16bf load_b_frag(const bf16_t* row) {
  v16bf b;
#pragma unroll
  for (int j = 0; j < 16; ++j) b[j] = row[j];
  return b;
}

// 32x64 register-blocked tile: D = A(MxK row-major bf16) x Bt^T,
// Bt is [n][K] bf16 (K-contiguous). 8 WMMAs per 32-deep k-step.
template<int KDIM>
__device__ __forceinline__ void gemm_tile_2x4(const bf16_t* __restrict__ A,
                                              const bf16_t* __restrict__ Bt,
                                              int ldb, int m0, int n0,
                                              int lm, int hlf, v8f acc[2][4]) {
  v8f zero = {};
#pragma unroll
  for (int mi = 0; mi < 2; ++mi)
#pragma unroll
    for (int ni = 0; ni < 4; ++ni) acc[mi][ni] = zero;
  const bf16_t* arow0 = A + (size_t)(m0 + lm) * KDIM;
  const bf16_t* arow1 = arow0 + (size_t)16 * KDIM;
  const bf16_t* brow  = Bt + (size_t)(n0 + lm) * ldb + hlf * 16;
#pragma unroll 2
  for (int k0 = 0; k0 < KDIM; k0 += 32) {
    // Prefetch next k-step (speculative; OOB prefetches are dropped).
    __builtin_prefetch(arow0 + k0 + 32, 0, 0);
    __builtin_prefetch(brow + k0 + 32, 0, 0);
    v16bf a0 = load_a_frag(arow0 + k0, hlf);
    v16bf a1 = load_a_frag(arow1 + k0, hlf);
    v16bf b0 = load_b_frag(brow + k0);
    v16bf b1 = load_b_frag(brow + (size_t)16 * ldb + k0);
    v16bf b2 = load_b_frag(brow + (size_t)32 * ldb + k0);
    v16bf b3 = load_b_frag(brow + (size_t)48 * ldb + k0);
    acc[0][0] = wmma_bf16(a0, b0, acc[0][0]);
    acc[0][1] = wmma_bf16(a0, b1, acc[0][1]);
    acc[0][2] = wmma_bf16(a0, b2, acc[0][2]);
    acc[0][3] = wmma_bf16(a0, b3, acc[0][3]);
    acc[1][0] = wmma_bf16(a1, b0, acc[1][0]);
    acc[1][1] = wmma_bf16(a1, b1, acc[1][1]);
    acc[1][2] = wmma_bf16(a1, b2, acc[1][2]);
    acc[1][3] = wmma_bf16(a1, b3, acc[1][3]);
  }
}

// ------------------------- GEMM wrapper kernels ----------------------------

// Out transposed bf16: Ot[b][n][m] (ldo columns). Used for qkv projections.
// Grid: BATCH*(M/32)*16 waves, 8 waves/block.
template<int M, int KDIM>
__global__ __launch_bounds__(256) void k_gemm_bf16T(const bf16_t* __restrict__ A,
                                                    const bf16_t* __restrict__ Bt,
                                                    bf16_t* __restrict__ Ot,
                                                    int ldb, int ldo) {
  int wid  = blockIdx.x * 8 + (threadIdx.x >> 5);
  int lane = threadIdx.x & 31;
  int lm   = lane & 15;
  int hlf  = lane >> 4;
  int nt = wid & 15;                 // 16 n-blocks of 64
  int mt = (wid >> 4) % (M / 32);
  int b  = wid / (16 * (M / 32));
  v8f acc[2][4];
  gemm_tile_2x4<KDIM>(A, Bt + (size_t)b * NPIX * ldb, ldb, mt * 32, nt * 64,
                      lm, hlf, acc);
#pragma unroll
  for (int mi = 0; mi < 2; ++mi)
#pragma unroll
    for (int ni = 0; ni < 4; ++ni) {
      bf16_t* orow = Ot + ((size_t)b * NPIX + nt * 64 + ni * 16 + lm) * ldo +
                     mt * 32 + mi * 16 + hlf * 8;
#pragma unroll
      for (int r = 0; r < 8; ++r) orow[r] = (bf16_t)acc[mi][ni][r];
    }
}

// Out f32 row-major O[b][m][n] (+bias[m], optional residual add). Used for
// proj, fc1, fc2.
template<int M, int KDIM, bool RES>
__global__ __launch_bounds__(256) void k_gemm_f32(const bf16_t* __restrict__ A,
                                                  const bf16_t* __restrict__ Bt,
                                                  const float* __restrict__ bias,
                                                  const float* __restrict__ res,
                                                  float* __restrict__ O, int ldb) {
  int wid  = blockIdx.x * 8 + (threadIdx.x >> 5);
  int lane = threadIdx.x & 31;
  int lm   = lane & 15;
  int hlf  = lane >> 4;
  int nt = wid & 15;
  int mt = (wid >> 4) % (M / 32);
  int b  = wid / (16 * (M / 32));
  v8f acc[2][4];
  gemm_tile_2x4<KDIM>(A, Bt + (size_t)b * NPIX * ldb, ldb, mt * 32, nt * 64,
                      lm, hlf, acc);
#pragma unroll
  for (int mi = 0; mi < 2; ++mi)
#pragma unroll
    for (int r = 0; r < 8; ++r) {
      int m = mt * 32 + mi * 16 + hlf * 8 + r;
      float bb = bias[m];
#pragma unroll
      for (int ni = 0; ni < 4; ++ni) {
        int n = nt * 64 + ni * 16 + lm;
        size_t idx = ((size_t)b * M + m) * NPIX + n;
        float v = acc[mi][ni][r] + bb;
        if (RES) v += res[idx];
        O[idx] = v;
      }
    }
}

// ------------------------- attention kernels -------------------------------

// Global attention: one wave per (b, head, 16-query-row tile).
// 3 sweeps: (1) S=Q^T K with online row max/sum, (2) recompute S -> normalized
// bf16 P in LDS, (3) O = P x V. All GEMMs use WMMA bf16.
__global__ __launch_bounds__(32) void k_ga_attn(const bf16_t* __restrict__ qkv_t,
                                                bf16_t* __restrict__ out_t) {
  __shared__ bf16_t pb[16 * NPIX];  // 32 KB probability tile
  int wid = blockIdx.x;
  int qt = wid & 63;
  int h  = (wid >> 6) & 7;
  int b  = wid >> 9;
  int lane = threadIdx.x & 31;
  int lm   = lane & 15;
  int hlf  = lane >> 4;
  const bf16_t* base = qkv_t + (size_t)b * NPIX * QKV_LD;
  const float scale = 0.17677669529663687f;  // 1/sqrt(32)
  int q0 = qt * 16;

  // Q fragment is reused across all key tiles.
  v16bf aq = load_a_frag(base + (size_t)(q0 + lm) * QKV_LD + h * HDIM, hlf);

  float M[8], L[8];
#pragma unroll
  for (int r = 0; r < 8; ++r) { M[r] = -3e38f; L[r] = 0.f; }

  // Pass 1: online row max / sum over 64 key tiles of 16.
  for (int kt = 0; kt < 64; ++kt) {
    const bf16_t* krow =
        base + (size_t)(kt * 16 + lm) * QKV_LD + CHANS + h * HDIM + hlf * 16;
    v16bf bk = load_b_frag(krow);
    v8f s = {};
    s = wmma_bf16(aq, bk, s);
#pragma unroll
    for (int r = 0; r < 8; ++r) {
      float v = s[r] * scale;
      float tm = v;
      tm = fmaxf(tm, __shfl_xor(tm, 1, 32));
      tm = fmaxf(tm, __shfl_xor(tm, 2, 32));
      tm = fmaxf(tm, __shfl_xor(tm, 4, 32));
      tm = fmaxf(tm, __shfl_xor(tm, 8, 32));
      float te = __expf(v - tm);
      te += __shfl_xor(te, 1, 32);
      te += __shfl_xor(te, 2, 32);
      te += __shfl_xor(te, 4, 32);
      te += __shfl_xor(te, 8, 32);
      float Mn = fmaxf(M[r], tm);
      L[r] = L[r] * __expf(M[r] - Mn) + te * __expf(tm - Mn);
      M[r] = Mn;
    }
  }
  float inv[8];
#pragma unroll
  for (int r = 0; r < 8; ++r) inv[r] = 1.0f / L[r];

  // Pass 2: recompute scores, write normalized bf16 probabilities to LDS.
  for (int kt = 0; kt < 64; ++kt) {
    const bf16_t* krow =
        base + (size_t)(kt * 16 + lm) * QKV_LD + CHANS + h * HDIM + hlf * 16;
    v16bf bk = load_b_frag(krow);
    v8f s = {};
    s = wmma_bf16(aq, bk, s);
#pragma unroll
    for (int r = 0; r < 8; ++r) {
      float p = __expf(s[r] * scale - M[r]) * inv[r];
      pb[(hlf * 8 + r) * NPIX + kt * 16 + lm] = (bf16_t)p;
    }
  }
  __syncthreads();

  // Pass 3: O = P x V over K=1024 keys (32 WMMA steps), two d-column tiles.
  v8f o0 = {}, o1 = {};
  const bf16_t* vbase = base + 2 * CHANS + h * HDIM;
  for (int kt = 0; kt < 32; ++kt) {
    v16bf ap = load_a_frag(pb + lm * NPIX + kt * 32, hlf);
    v16bf bv0, bv1;
#pragma unroll
    for (int j = 0; j < 8; ++j) {
      int kk = kt * 32 + hlf * 16 + 2 * j;
      bv0[2 * j]     = vbase[(size_t)kk * QKV_LD + lm];
      bv0[2 * j + 1] = vbase[(size_t)(kk + 1) * QKV_LD + lm];
      bv1[2 * j]     = vbase[(size_t)kk * QKV_LD + 16 + lm];
      bv1[2 * j + 1] = vbase[(size_t)(kk + 1) * QKV_LD + 16 + lm];
    }
    o0 = wmma_bf16(ap, bv0, o0);
    o1 = wmma_bf16(ap, bv1, o1);
  }
  bf16_t* obase = out_t + ((size_t)b * NPIX + q0) * CHANS + h * HDIM;
#pragma unroll
  for (int r = 0; r < 8; ++r) {
    obase[(size_t)(hlf * 8 + r) * CHANS + lm]      = (bf16_t)o0[r];
    obase[(size_t)(hlf * 8 + r) * CHANS + 16 + lm] = (bf16_t)o1[r];
  }
}

// Window attention (8x8 windows, 64 tokens): VALU online softmax, K/V in LDS.
// FLOP-light (~0.5 GF) so no WMMA needed here.
__global__ __launch_bounds__(64) void k_win_attn(const bf16_t* __restrict__ qkv_t,
                                                 bf16_t* __restrict__ out_t) {
  __shared__ float Ks[64][HDIM];
  __shared__ float Vs[64][HDIM];
  int blk = blockIdx.x;          // b*16*8 + win*8 + h
  int h = blk & 7;
  int w = (blk >> 3) & 15;
  int b = blk >> 7;
  int wy = w >> 2, wx = w & 3;
  int t = threadIdx.x;           // token in window
  int n = (wy * 8 + (t >> 3)) * SIDE + (wx * 8 + (t & 7));
  const bf16_t* row = qkv_t + ((size_t)b * NPIX + n) * QKV_LD + h * HDIM;
  float q[HDIM];
#pragma unroll
  for (int d = 0; d < HDIM; ++d) {
    q[d]     = (float)row[d];
    Ks[t][d] = (float)row[CHANS + d];
    Vs[t][d] = (float)row[2 * CHANS + d];
  }
  __syncthreads();
  const float scale = 0.17677669529663687f;
  float M = -3e38f, L = 0.f, acc[HDIM];
#pragma unroll
  for (int d = 0; d < HDIM; ++d) acc[d] = 0.f;
  for (int m = 0; m < 64; ++m) {
    float s = 0.f;
#pragma unroll
    for (int d = 0; d < HDIM; ++d) s += q[d] * Ks[m][d];
    s *= scale;
    float Mn = fmaxf(M, s);
    float corr = __expf(M - Mn);
    float wgt  = __expf(s - Mn);
    L = L * corr + wgt;
#pragma unroll
    for (int d = 0; d < HDIM; ++d) acc[d] = acc[d] * corr + wgt * Vs[m][d];
    M = Mn;
  }
  float invL = 1.0f / L;
  bf16_t* orow = out_t + ((size_t)b * NPIX + n) * CHANS + h * HDIM;
#pragma unroll
  for (int d = 0; d < HDIM; ++d) orow[d] = (bf16_t)(acc[d] * invL);
}

// ------------------------- elementwise / reductions ------------------------

__global__ __launch_bounds__(256) void k_cvt(const float* __restrict__ s,
                                             bf16_t* __restrict__ d, int n) {
  int i = blockIdx.x * 256 + threadIdx.x;
  if (i < n) d[i] = (bf16_t)s[i];
}

// Per (b,c) row over 1024 pixels: mean, rsqrt(var+eps), optional s=mean(xn).
__global__ __launch_bounds__(256) void k_rowstats(const float* __restrict__ X,
                                                  float* __restrict__ mo,
                                                  float* __restrict__ ro,
                                                  float* __restrict__ so, int ws) {
  __shared__ float sa[256], sb[256];
  int row = blockIdx.x;
  const float* p = X + (size_t)row * NPIX;
  float a = 0.f, bq = 0.f;
  for (int i = threadIdx.x; i < NPIX; i += 256) {
    float v = p[i];
    a += v;
    bq += v * v;
  }
  sa[threadIdx.x] = a;
  sb[threadIdx.x] = bq;
  __syncthreads();
  for (int st = 128; st; st >>= 1) {
    if (threadIdx.x < st) {
      sa[threadIdx.x] += sa[threadIdx.x + st];
      sb[threadIdx.x] += sb[threadIdx.x + st];
    }
    __syncthreads();
  }
  if (threadIdx.x == 0) {
    float mean = sa[0] * (1.0f / NPIX);
    float var  = sb[0] * (1.0f / NPIX) - mean * mean;
    float rr   = rsqrtf(var + 1e-5f);
    mo[row] = mean;
    ro[row] = rr;
    if (ws) so[row] = (sa[0] * (1.0f / NPIX) - mean) * rr;  // exactly ~0
  }
}

// Write normalized bf16 transposed: dst[b][n][c] = (x[b][c][n]-m)*r
__global__ __launch_bounds__(256) void k_writeT(const float* __restrict__ X,
                                                const float* __restrict__ m,
                                                const float* __restrict__ r,
                                                bf16_t* __restrict__ dst) {
  size_t i = (size_t)blockIdx.x * 256 + threadIdx.x;  // over B*C*N
  int n = (int)(i & (NPIX - 1));
  int c = (int)((i >> 10) & (CHANS - 1));
  int b = (int)(i >> 18);
  int row = b * CHANS + c;
  float v = (X[i] - m[row]) * r[row];
  dst[((size_t)b * NPIX + n) * CHANS + c] = (bf16_t)v;
}

__global__ __launch_bounds__(256) void k_add3(const float* __restrict__ x,
                                              const float* __restrict__ yw,
                                              const float* __restrict__ yg,
                                              float* __restrict__ o) {
  size_t i = (size_t)blockIdx.x * 256 + threadIdx.x;
  o[i] = x[i] + yw[i] + yg[i];
}

// Channel attention MLP: scale[b][c] = sigmoid(W2 . relu(W1 . s[b]))
__global__ __launch_bounds__(256) void k_ca_mlp(const float* __restrict__ s,
                                                const float* __restrict__ w1,
                                                const float* __restrict__ w2,
                                                float* __restrict__ scale) {
  __shared__ float hid[16];
  int b = blockIdx.x;
  if (threadIdx.x < 16) {
    float a = 0.f;
    for (int k = 0; k < CHANS; ++k) a += w1[threadIdx.x * CHANS + k] * s[b * CHANS + k];
    hid[threadIdx.x] = fmaxf(a, 0.f);
  }
  __syncthreads();
  int c = threadIdx.x;
  float a = 0.f;
#pragma unroll
  for (int j = 0; j < 16; ++j) a += w2[c * 16 + j] * hid[j];
  scale[b * CHANS + c] = 1.0f / (1.0f + __expf(-a));
}

__global__ __launch_bounds__(256) void k_x3(const float* __restrict__ x2,
                                            const float* __restrict__ m2,
                                            const float* __restrict__ r2,
                                            const float* __restrict__ scale,
                                            float* __restrict__ x3) {
  size_t i = (size_t)blockIdx.x * 256 + threadIdx.x;
  int c = (int)((i >> 10) & (CHANS - 1));
  int b = (int)(i >> 18);
  int row = b * CHANS + c;
  float v = x2[i];
  x3[i] = v + scale[row] * (v - m2[row]) * r2[row];
}

__global__ __launch_bounds__(256) void k_rowmean(const float* __restrict__ X,
                                                 float* __restrict__ mo) {
  __shared__ float sa[256];
  int row = blockIdx.x;
  const float* p = X + (size_t)row * NPIX;
  float a = 0.f;
  for (int i = threadIdx.x; i < NPIX; i += 256) a += p[i];
  sa[threadIdx.x] = a;
  __syncthreads();
  for (int st = 128; st; st >>= 1) {
    if (threadIdx.x < st) sa[threadIdx.x] += sa[threadIdx.x + st];
    __syncthreads();
  }
  if (threadIdx.x == 0) mo[row] = sa[0] * (1.0f / NPIX);
}

// Broadcast g = mean(h1) into z_t[b][n][1024+o]
__global__ __launch_bounds__(256) void k_ztg(const float* __restrict__ gmean,
                                             bf16_t* __restrict__ z_t) {
  size_t i = (size_t)blockIdx.x * 256 + threadIdx.x;  // over B*N*HID
  int o = (int)(i & (HIDDIM - 1));
  int n = (int)((i >> 10) & (NPIX - 1));
  int b = (int)(i >> 20);
  z_t[((size_t)b * NPIX + n) * K2DIM + HIDDIM + o] = (bf16_t)gmean[b * HIDDIM + o];
}

// Depthwise KSxKS conv (SAME, zero pad) + exact GELU, write bf16 to z_t.
template<int KS>
__global__ __launch_bounds__(256) void k_dw(const float* __restrict__ h1,
                                            const float* __restrict__ w,
                                            const float* __restrict__ bias,
                                            bf16_t* __restrict__ z_t,
                                            int ch_off, int z_off) {
  size_t i = (size_t)blockIdx.x * 256 + threadIdx.x;  // over B*512*N
  int n = (int)(i & (NPIX - 1));
  int c = (int)((i >> 10) & 511);
  int b = (int)(i >> 19);
  int y = n >> 5, x = n & 31;
  const float* plane = h1 + ((size_t)b * HIDDIM + ch_off + c) * NPIX;
  const float* wk = w + c * KS * KS;
  const int R = KS / 2;
  float acc = bias[c];
#pragma unroll
  for (int dy = -R; dy <= R; ++dy) {
#pragma unroll
    for (int dx = -R; dx <= R; ++dx) {
      int yy = y + dy, xx = x + dx;
      if (yy >= 0 && yy < SIDE && xx >= 0 && xx < SIDE)
        acc += plane[yy * SIDE + xx] * wk[(dy + R) * KS + (dx + R)];
    }
  }
  float g = 0.5f * acc * (1.0f + erff(acc * 0.70710678118654752f));
  z_t[((size_t)b * NPIX + n) * K2DIM + z_off + c] = (bf16_t)g;
}

// ---------------------------------------------------------------------------

extern "C" void kernel_launch(void* const* d_in, const int* in_sizes, int n_in,
                              void* d_out, int out_size, void* d_ws, size_t ws_size,
                              hipStream_t stream) {
  (void)in_sizes; (void)n_in; (void)out_size; (void)ws_size;
  const float* x      = (const float*)d_in[0];
  const float* wa_qkv = (const float*)d_in[1];
  const float* wa_pw  = (const float*)d_in[2];
  const float* wa_pb  = (const float*)d_in[3];
  const float* ga_qkv = (const float*)d_in[4];
  const float* ga_pw  = (const float*)d_in[5];
  const float* ga_pb  = (const float*)d_in[6];
  const float* ca_w1  = (const float*)d_in[7];
  const float* ca_w2  = (const float*)d_in[8];
  const float* fc1_w  = (const float*)d_in[9];
  const float* fc1_b  = (const float*)d_in[10];
  const float* dw3_w  = (const float*)d_in[11];
  const float* dw3_b  = (const float*)d_in[12];
  const float* dw5_w  = (const float*)d_in[13];
  const float* dw5_b  = (const float*)d_in[14];
  const float* fc2_w  = (const float*)d_in[15];
  const float* fc2_b  = (const float*)d_in[16];
  float* out = (float*)d_out;

  size_t off = 0;
  auto alloc = [&](size_t bytes) -> char* {
    off = (off + 255) & ~(size_t)255;
    char* p = (char*)d_ws + off;
    off += bytes;
    return p;
  };

  bf16_t* wa_qkv_bf = (bf16_t*)alloc((size_t)QKV_LD * CHANS * 2);
  bf16_t* ga_qkv_bf = (bf16_t*)alloc((size_t)QKV_LD * CHANS * 2);
  bf16_t* wa_pw_bf  = (bf16_t*)alloc((size_t)CHANS * CHANS * 2);
  bf16_t* ga_pw_bf  = (bf16_t*)alloc((size_t)CHANS * CHANS * 2);
  bf16_t* fc1_bf    = (bf16_t*)alloc((size_t)HIDDIM * CHANS * 2);
  bf16_t* fc2_bf    = (bf16_t*)alloc((size_t)CHANS * K2DIM * 2);
  bf16_t* xnT       = (bf16_t*)alloc((size_t)BATCH * NPIX * CHANS * 2);
  // qkvw_t..og_t are contiguous (all sizes are 256-multiples); h1 aliases them.
  bf16_t* qkvw_t    = (bf16_t*)alloc((size_t)BATCH * NPIX * QKV_LD * 2);
  bf16_t* qkvg_t    = (bf16_t*)alloc((size_t)BATCH * NPIX * QKV_LD * 2);
  bf16_t* ow_t      = (bf16_t*)alloc((size_t)BATCH * NPIX * CHANS * 2);
  bf16_t* og_t      = (bf16_t*)alloc((size_t)BATCH * NPIX * CHANS * 2);
  float*  yw        = (float*)alloc((size_t)BATCH * CHANS * NPIX * 4);
  float*  yg        = (float*)alloc((size_t)BATCH * CHANS * NPIX * 4);
  float*  x2        = (float*)alloc((size_t)BATCH * CHANS * NPIX * 4);
  float*  x3        = (float*)alloc((size_t)BATCH * CHANS * NPIX * 4);
  bf16_t* z_t       = (bf16_t*)alloc((size_t)BATCH * NPIX * K2DIM * 2);
  float*  gmean     = (float*)alloc((size_t)BATCH * HIDDIM * 4);
  float*  m1 = (float*)alloc(BATCH * CHANS * 4);
  float*  r1 = (float*)alloc(BATCH * CHANS * 4);
  float*  m2 = (float*)alloc(BATCH * CHANS * 4);
  float*  r2 = (float*)alloc(BATCH * CHANS * 4);
  float*  m3 = (float*)alloc(BATCH * CHANS * 4);
  float*  r3 = (float*)alloc(BATCH * CHANS * 4);
  float*  svec   = (float*)alloc(BATCH * CHANS * 4);
  float*  scalev = (float*)alloc(BATCH * CHANS * 4);
  // Aliases: h1 (32MB) over the dead qkv/attention-out region; xn3T over xnT.
  float*  h1   = (float*)qkvw_t;
  bf16_t* xn3T = xnT;

  // --- weight conversion to bf16 ---
  k_cvt<<<(QKV_LD * CHANS + 255) / 256, 256, 0, stream>>>(wa_qkv, wa_qkv_bf, QKV_LD * CHANS);
  k_cvt<<<(QKV_LD * CHANS + 255) / 256, 256, 0, stream>>>(ga_qkv, ga_qkv_bf, QKV_LD * CHANS);
  k_cvt<<<(CHANS * CHANS + 255) / 256, 256, 0, stream>>>(wa_pw, wa_pw_bf, CHANS * CHANS);
  k_cvt<<<(CHANS * CHANS + 255) / 256, 256, 0, stream>>>(ga_pw, ga_pw_bf, CHANS * CHANS);
  k_cvt<<<(HIDDIM * CHANS + 255) / 256, 256, 0, stream>>>(fc1_w, fc1_bf, HIDDIM * CHANS);
  k_cvt<<<(CHANS * K2DIM + 255) / 256, 256, 0, stream>>>(fc2_w, fc2_bf, CHANS * K2DIM);

  const int EW = BATCH * CHANS * NPIX / 256;  // 8192 elementwise blocks
  // --- stage 1: inorm -> xnT, qkv projections (WMMA) ---
  k_rowstats<<<BATCH * CHANS, 256, 0, stream>>>(x, m1, r1, nullptr, 0);
  k_writeT<<<EW, 256, 0, stream>>>(x, m1, r1, xnT);
  k_gemm_bf16T<QKV_LD, CHANS><<<BATCH * (QKV_LD / 32) * 16 / 8, 256, 0, stream>>>(
      wa_qkv_bf, xnT, qkvw_t, CHANS, QKV_LD);
  k_gemm_bf16T<QKV_LD, CHANS><<<BATCH * (QKV_LD / 32) * 16 / 8, 256, 0, stream>>>(
      ga_qkv_bf, xnT, qkvg_t, CHANS, QKV_LD);

  // --- attention ---
  k_win_attn<<<BATCH * 16 * NHEADS, 64, 0, stream>>>(qkvw_t, ow_t);
  k_ga_attn<<<BATCH * NHEADS * 64, 32, 0, stream>>>(qkvg_t, og_t);

  // --- output projections (WMMA) + residual ---
  k_gemm_f32<CHANS, CHANS, false><<<BATCH * (CHANS / 32) * 16 / 8, 256, 0, stream>>>(
      wa_pw_bf, ow_t, wa_pb, nullptr, yw, CHANS);
  k_gemm_f32<CHANS, CHANS, false><<<BATCH * (CHANS / 32) * 16 / 8, 256, 0, stream>>>(
      ga_pw_bf, og_t, ga_pb, nullptr, yg, CHANS);
  k_add3<<<EW, 256, 0, stream>>>(x, yw, yg, x2);

  // --- channel attention ---
  k_rowstats<<<BATCH * CHANS, 256, 0, stream>>>(x2, m2, r2, svec, 1);
  k_ca_mlp<<<BATCH, 256, 0, stream>>>(svec, ca_w1, ca_w2, scalev);
  k_x3<<<EW, 256, 0, stream>>>(x2, m2, r2, scalev, x3);

  // --- ConvFFN: inorm -> fc1 (WMMA) -> dw3/dw5+gelu + global mean -> fc2 ---
  k_rowstats<<<BATCH * CHANS, 256, 0, stream>>>(x3, m3, r3, nullptr, 0);
  k_writeT<<<EW, 256, 0, stream>>>(x3, m3, r3, xn3T);
  k_gemm_f32<HIDDIM, CHANS, false><<<BATCH * (HIDDIM / 32) * 16 / 8, 256, 0, stream>>>(
      fc1_bf, xn3T, fc1_b, nullptr, h1, CHANS);
  k_rowmean<<<BATCH * HIDDIM, 256, 0, stream>>>(h1, gmean);
  k_ztg<<<BATCH * NPIX * HIDDIM / 256, 256, 0, stream>>>(gmean, z_t);
  k_dw<3><<<BATCH * 512 * NPIX / 256, 256, 0, stream>>>(h1, dw3_w, dw3_b, z_t, 0, 0);
  k_dw<5><<<BATCH * 512 * NPIX / 256, 256, 0, stream>>>(h1, dw5_w, dw5_b, z_t, 512, 512);
  k_gemm_f32<CHANS, K2DIM, true><<<BATCH * (CHANS / 32) * 16 / 8, 256, 0, stream>>>(
      fc2_bf, z_t, fc2_b, x3, out, K2DIM);
}
